// HeadV3_18468359373430
// MI455X (gfx1250) — compile-verified, matrix operand order checked by reading, not run on hardware
//
#include <hip/hip_runtime.h>
#include <hip/hip_bf16.h>

// ---------------------------------------------------------------------------
// Fused causal single-head attention for MI455X (gfx1250, wave32, WMMA).
//   Pass 1: qkv_proj  — x[B*T,C] fp32 @ {Wq,Wk,Wv}[C,D] fp32 -> f16 Q,K (row
//           major) and V^T ([B,D,T]) in workspace, via v_wmma_f32_16x16x32_f16.
//   Pass 2: attn      — flash-attention (online softmax), causal, scale C^-0.5,
//           fp32 accumulators, f16 WMMA for S=Q K^T and O += P V.
// ---------------------------------------------------------------------------

#define B_ 8
#define T_ 2048
#define C_ 1024
#define D_ 64

typedef __attribute__((ext_vector_type(16))) _Float16 v16h;
typedef __attribute__((ext_vector_type(8)))  _Float16 v8h;
typedef __attribute__((ext_vector_type(8)))  float    v8f;

static __device__ __forceinline__ v16h cat16(v8h lo, v8h hi) {
    return __builtin_shufflevector(lo, hi, 0,1,2,3,4,5,6,7,8,9,10,11,12,13,14,15);
}

static __device__ __forceinline__ v8f wmma_f16(v16h a, v16h b, v8f c) {
    return __builtin_amdgcn_wmma_f32_16x16x32_f16(
        /*neg_a=*/false, a, /*neg_b=*/false, b,
        /*c_mod=*/(short)0, c, /*reuse_a=*/false, /*reuse_b=*/false);
}

// ---------------------------------------------------------------------------
// Pass 1: fused QKV projection. 256 threads = 8 waves; each wave computes a
// 16-row x 64-col tile for each of q,k,v. Block handles 128 rows of B*T.
// ---------------------------------------------------------------------------
__global__ __launch_bounds__(256) void qkv_proj_kernel(
    const float* __restrict__ x,     // [B*T, C]
    const float* __restrict__ Wq,    // [C, D]
    const float* __restrict__ Wk,    // [C, D]
    const float* __restrict__ Wv,    // [C, D]
    _Float16* __restrict__ Qh,       // [B*T, D]
    _Float16* __restrict__ Kh,       // [B*T, D]
    _Float16* __restrict__ Vt)       // [B, D, T]  (transposed V)
{
    // Transposed f16 W slices: [mat][n][k], k padded 32->40 so every
    // 8-halfword run stays 16B aligned (row stride 80B).
    __shared__ __align__(16) _Float16 Wlds[3][64][40];

    const int tid   = threadIdx.x;
    const int wave  = tid >> 5;
    const int lane  = tid & 31;
    const int lhalf = lane >> 4;   // K-half selector (A/B layout)
    const int l16   = lane & 15;

    const int rbase = blockIdx.x * 128;            // first row of block
    const int arow  = rbase + wave * 16 + l16;     // this lane's A row

    v8f acc[3][4] = {};                            // [mat][ntile] 16x16 f32

    for (int kc = 0; kc < C_ / 32; ++kc) {
        __syncthreads();
        // Cooperative load: 3 x (32 x 64) fp32 -> f16, transposed into LDS.
        #pragma unroll
        for (int i = 0; i < 24; ++i) {
            const int m = i >> 3;                  // matrix id (const per i)
            const float* W = (m == 0) ? Wq : (m == 1) ? Wk : Wv;
            int idx = tid + i * 256;
            int rem = idx & 2047;
            int kk  = rem >> 6;                    // 0..31
            int n   = rem & 63;                    // 0..63 (coalesced)
            Wlds[m][n][kk] = (_Float16)W[(size_t)(kc * 32 + kk) * D_ + n];
        }
        __syncthreads();

        // A tile: x rows, fp32 b128 loads, cvt -> f16, ISA A-layout.
        const float* xp = x + (size_t)arow * C_ + kc * 32 + lhalf * 8;
        const float4 x0 = *(const float4*)(xp + 0);
        const float4 x1 = *(const float4*)(xp + 4);
        const float4 x2 = *(const float4*)(xp + 16);
        const float4 x3 = *(const float4*)(xp + 20);
        v16h a;
        a[0]=(_Float16)x0.x; a[1]=(_Float16)x0.y; a[2]=(_Float16)x0.z; a[3]=(_Float16)x0.w;
        a[4]=(_Float16)x1.x; a[5]=(_Float16)x1.y; a[6]=(_Float16)x1.z; a[7]=(_Float16)x1.w;
        a[8]=(_Float16)x2.x; a[9]=(_Float16)x2.y; a[10]=(_Float16)x2.z; a[11]=(_Float16)x2.w;
        a[12]=(_Float16)x3.x; a[13]=(_Float16)x3.y; a[14]=(_Float16)x3.z; a[15]=(_Float16)x3.w;

        #pragma unroll
        for (int m = 0; m < 3; ++m) {
            #pragma unroll
            for (int nt = 0; nt < 4; ++nt) {
                const _Float16* wp = &Wlds[m][nt * 16 + l16][lhalf * 8];
                v8h blo = *(const v8h*)(wp);
                v8h bhi = *(const v8h*)(wp + 16);
                acc[m][nt] = wmma_f16(a, cat16(blo, bhi), acc[m][nt]);
            }
        }
    }

    // Store results. C/D layout: lanes 0-15 -> M=r, lanes 16-31 -> M=r+8.
    const int bt0 = rbase + wave * 16;
    const int b   = bt0 / T_;
    #pragma unroll
    for (int nt = 0; nt < 4; ++nt) {
        #pragma unroll
        for (int r = 0; r < 8; ++r) {
            const int M = r + lhalf * 8;
            const int t = bt0 + M;
            const int d = nt * 16 + l16;
            Qh[(size_t)t * D_ + d] = (_Float16)acc[0][nt][r];
            Kh[(size_t)t * D_ + d] = (_Float16)acc[1][nt][r];
            Vt[((size_t)b * D_ + d) * T_ + (t - b * T_)] = (_Float16)acc[2][nt][r];
        }
    }
}

// ---------------------------------------------------------------------------
// Pass 2: flash attention. 128 threads = 4 waves; each wave owns 16 queries.
// Key tiles of 32; online softmax with per-row (m, l) state.
// ---------------------------------------------------------------------------
#define AW 4  // waves per block

__global__ __launch_bounds__(128) void attn_kernel(
    const _Float16* __restrict__ Qh,   // [B*T, D]
    const _Float16* __restrict__ Kh,   // [B*T, D]
    const _Float16* __restrict__ Vt,   // [B, D, T]
    float* __restrict__ out)           // [B*T, D]
{
    __shared__ __align__(16) _Float16 Plds[AW][16][40];  // per-wave P staging

    const int tid   = threadIdx.x;
    const int wave  = tid >> 5;
    const int lane  = tid & 31;
    const int lhalf = lane >> 4;
    const int l16   = lane & 15;

    const int qblocks = T_ / (16 * AW);
    const int b   = blockIdx.x / qblocks;
    const int qb  = blockIdx.x % qblocks;
    const int qlo = qb * (16 * AW) + wave * 16;    // first query of this wave
    const int qhi = qlo + 15;

    const float scale  = 0.03125f;                 // 1024^-0.5
    const float NEGBIG = -1.0e30f;

    // Q A-tiles (2 chunks of K=32 over D=64), resident in registers.
    v16h qa[2];
    {
        const _Float16* qp = Qh + (size_t)(b * T_ + qlo + l16) * D_;
        #pragma unroll
        for (int c = 0; c < 2; ++c) {
            v8h lo = *(const v8h*)(qp + c * 32 + lhalf * 8);
            v8h hi = *(const v8h*)(qp + c * 32 + 16 + lhalf * 8);
            qa[c] = cat16(lo, hi);
        }
    }

    v8f o[4] = {};
    float mrow[8], lrow[8];
    #pragma unroll
    for (int r = 0; r < 8; ++r) { mrow[r] = NEGBIG; lrow[r] = 0.0f; }

    for (int kbase = 0; kbase <= qhi; kbase += 32) {
        // ---- S = Q K^T for two 16-key subtiles -------------------------
        v8f s[2];
        #pragma unroll
        for (int sub = 0; sub < 2; ++sub) {
            int krow = kbase + sub * 16 + l16;
            if (krow > T_ - 1) krow = T_ - 1;      // clamp; masked anyway
            const _Float16* kp = Kh + (size_t)(b * T_ + krow) * D_;
            v8f sa = {};
            #pragma unroll
            for (int c = 0; c < 2; ++c) {
                v8h lo = *(const v8h*)(kp + c * 32 + lhalf * 8);
                v8h hi = *(const v8h*)(kp + c * 32 + 16 + lhalf * 8);
                sa = wmma_f16(qa[c], cat16(lo, hi), sa);
            }
            s[sub] = sa;
        }

        // ---- scale + causal mask --------------------------------------
        const bool need_mask = (kbase + 31) > qlo;
        #pragma unroll
        for (int sub = 0; sub < 2; ++sub) {
            #pragma unroll
            for (int r = 0; r < 8; ++r) {
                float sv = s[sub][r] * scale;
                if (need_mask) {
                    const int key = kbase + sub * 16 + l16;
                    const int qry = qlo + r + lhalf * 8;
                    if (key > qry) sv = NEGBIG;
                }
                s[sub][r] = sv;
            }
        }

        // ---- online softmax (row reductions within 16-lane halves) ----
        #pragma unroll
        for (int r = 0; r < 8; ++r) {
            float mx = fmaxf(s[0][r], s[1][r]);
            mx = fmaxf(mx, __shfl_xor(mx, 1));
            mx = fmaxf(mx, __shfl_xor(mx, 2));
            mx = fmaxf(mx, __shfl_xor(mx, 4));
            mx = fmaxf(mx, __shfl_xor(mx, 8));
            mx = fmaxf(mx, mrow[r]);
            const float alpha = __expf(mrow[r] - mx);
            mrow[r] = mx;
            const float p0 = __expf(s[0][r] - mx);
            const float p1 = __expf(s[1][r] - mx);
            s[0][r] = p0; s[1][r] = p1;
            float ps = p0 + p1;
            ps += __shfl_xor(ps, 1);
            ps += __shfl_xor(ps, 2);
            ps += __shfl_xor(ps, 4);
            ps += __shfl_xor(ps, 8);
            lrow[r] = lrow[r] * alpha + ps;
            #pragma unroll
            for (int nt = 0; nt < 4; ++nt) o[nt][r] *= alpha;
        }

        // ---- re-lay P (C/D layout) -> A layout via per-wave LDS --------
        #pragma unroll
        for (int sub = 0; sub < 2; ++sub) {
            #pragma unroll
            for (int r = 0; r < 8; ++r) {
                Plds[wave][r + lhalf * 8][sub * 16 + l16] = (_Float16)s[sub][r];
            }
        }
        asm volatile("s_wait_dscnt 0x0" ::: "memory");  // same-wave LDS RAW
        v16h pa;
        {
            const _Float16* pp = &Plds[wave][l16][lhalf * 8];
            v8h lo = *(const v8h*)(pp);
            v8h hi = *(const v8h*)(pp + 16);
            pa = cat16(lo, hi);
        }
        asm volatile("" ::: "memory");  // keep next-iter P stores after reads

        // ---- O += P V  (V^T gives contiguous B-tile loads) -------------
        int kt0 = kbase + lhalf * 8;
        int kt1 = kbase + 16 + lhalf * 8;
        if (kt0 > T_ - 8) kt0 = T_ - 8;  // clamped tails are multiplied by p=0
        if (kt1 > T_ - 8) kt1 = T_ - 8;
        #pragma unroll
        for (int nt = 0; nt < 4; ++nt) {
            const _Float16* vp = Vt + ((size_t)b * D_ + nt * 16 + l16) * T_;
            v8h lo = *(const v8h*)(vp + kt0);
            v8h hi = *(const v8h*)(vp + kt1);
            o[nt] = wmma_f16(pa, cat16(lo, hi), o[nt]);
        }
    }

    // ---- epilogue: out = o / l ----------------------------------------
    #pragma unroll
    for (int r = 0; r < 8; ++r) {
        const float invl = 1.0f / lrow[r];
        const int t = qlo + r + lhalf * 8;
        #pragma unroll
        for (int nt = 0; nt < 4; ++nt) {
            out[(size_t)(b * T_ + t) * D_ + nt * 16 + l16] = o[nt][r] * invl;
        }
    }
}

// ---------------------------------------------------------------------------
extern "C" void kernel_launch(void* const* d_in, const int* in_sizes, int n_in,
                              void* d_out, int out_size, void* d_ws, size_t ws_size,
                              hipStream_t stream) {
    (void)in_sizes; (void)n_in; (void)out_size; (void)ws_size;
    const float* x  = (const float*)d_in[0];
    const float* Wk = (const float*)d_in[1];
    const float* Wq = (const float*)d_in[2];
    const float* Wv = (const float*)d_in[3];
    float* out = (float*)d_out;

    _Float16* Qh = (_Float16*)d_ws;                      // [B*T, D] f16
    _Float16* Kh = Qh + (size_t)B_ * T_ * D_;            // [B*T, D] f16
    _Float16* Vt = Kh + (size_t)B_ * T_ * D_;            // [B, D, T] f16
    // total workspace use: 3 * 8*2048*64 * 2B = 6 MB

    qkv_proj_kernel<<<dim3((B_ * T_) / 128), dim3(256), 0, stream>>>(
        x, Wq, Wk, Wv, Qh, Kh, Vt);

    attn_kernel<<<dim3(B_ * (T_ / (16 * AW))), dim3(32 * AW), 0, stream>>>(
        Qh, Kh, Vt, out);
}
